// MinkowskiUNet8_39883066311120
// MI455X (gfx1250) — matrix operand order repaired and checked by reading, not compiled, not run on hardware
//
#include <hip/hip_runtime.h>

// MI455X (gfx1250) implicit-GEMM 3D U-Net.
//  - V_WMMA_F32_16X16X32_F16, 32x16 output tile per wave (2 acc, shared B frag)
//  - Weight tiles DMA'd into LDS by the Tensor Data Mover (tensor_load_to_lds),
//    synchronized with s_wait_tensorcnt; TDM zero-fills K/N tail OOB.
//  - All layer shapes are template constants: no runtime division chains.
//  - BN folded to scale/bias in epilogue, ReLU fused, concats via dual input.

typedef __attribute__((ext_vector_type(16))) _Float16     v16h;
typedef __attribute__((ext_vector_type(8)))  float        v8f;
typedef __attribute__((ext_vector_type(4)))  unsigned int u32x4;
typedef __attribute__((ext_vector_type(8)))  int          i32x8;
typedef __attribute__((ext_vector_type(4)))  int          i32x4;

#define USE_TDM (__has_builtin(__builtin_amdgcn_tensor_load_to_lds))

template<int DI, int DO, int CIN1, int CIN2, int COUT, int KS, int STRIDE,
         int PADLO, int TRANSPOSED, int IN_RELU, int OUT_RELU, int FINAL>
__global__ __launch_bounds__(32)
void conv3d_wmma(const _Float16* __restrict__ in1,
                 const _Float16* __restrict__ in2,
                 const _Float16* __restrict__ wp,
                 const float* __restrict__ bn_g, const float* __restrict__ bn_b,
                 const float* __restrict__ bn_m, const float* __restrict__ bn_v,
                 _Float16* __restrict__ out, float* __restrict__ out_f32)
{
  constexpr int CIN  = CIN1 + CIN2;
  constexpr int TAPS = KS * KS * KS;
  constexpr int K    = TAPS * CIN;
  constexpr int M    = 2 * DO * DO * DO;

  const int lane  = threadIdx.x;        // wave32
  const int half  = lane >> 4;
  const int lan16 = lane & 15;
  const int mtile = blockIdx.x;
  const int ntile = blockIdx.y;

  __shared__ _Float16 As[32 * 32];      // 32 voxels x 32 K
  __shared__ _Float16 Bs[32 * 16];      // 32 K x 16 Cout (TDM destination)
  __shared__ int vb[32], vz[32], vy[32], vx[32];

  {                                     // decode this tile's 32 output voxels
    int vox = mtile * 32 + lane;
    if (vox >= M) vox = M - 1;          // clamp; masked at write
    int xq = vox % DO; int t = vox / DO;
    int yq = t % DO;   t /= DO;
    int zq = t % DO;   int bq = t / DO;
    vb[lane] = bq; vz[lane] = zq; vy[lane] = yq; vx[lane] = xq;
  }
  __syncthreads();

  v8f acc0 = {}, acc1 = {};

  for (int k0 = 0; k0 < K; k0 += 32) {
    // per-lane K decode: this lane stages column kk == lane of the A tile
    const int  kg     = k0 + lane;
    const bool kvalid = kg < K;
    const int  kc     = kvalid ? kg : 0;
    const int  tap    = kc / CIN;            // compile-time divisors
    const int  ci     = kc - tap * CIN;
    const int  kz     = tap / (KS * KS);
    const int  rr     = tap - kz * (KS * KS);
    const int  ky     = rr / KS;
    const int  kx     = rr - ky * KS;

#if USE_TDM
    // ---- Tensor Data Mover: DMA 32(K) x 16(N) weight tile into Bs.
    //      tensor_dim0/1 = remaining cols/rows -> OOB reads zero-fill tails.
    {
      const unsigned      ldsB = (unsigned)(uintptr_t)(&Bs[0]);
      const unsigned long ga =
          (unsigned long)(uintptr_t)(wp + (long)k0 * COUT + ntile * 16);
      u32x4 g0;
      g0.x = 1u;                                    // count=1, valid user D#
      g0.y = ldsB;                                  // lds_addr (bytes)
      g0.z = (unsigned)(ga & 0xffffffffu);          // global_addr[31:0]
      g0.w = (unsigned)((ga >> 32) & 0x01ffffffu)   // global_addr[56:32]
             | (2u << 30);                          // type = 2 ("image")
      const int td0 = COUT - ntile * 16;            // cols remaining
      const int td1 = K - k0;                       // rows remaining
      i32x8 g1;
      g1[0] = 1 << 16;                              // data_size = 2 bytes
      g1[1] = (td0 & 0xffff) << 16;                 // tensor_dim0[15:0]
      g1[2] = ((td0 >> 16) & 0xffff) | ((td1 & 0xffff) << 16);
      g1[3] = ((td1 >> 16) & 0xffff) | (16 << 16);  // tile_dim0 = 16
      g1[4] = 32;                                   // tile_dim1 = 32
      g1[5] = COUT;                                 // tensor_dim0_stride
      g1[6] = 0;
      g1[7] = 0;
      const i32x4 gz4 = {0, 0, 0, 0};
      const i32x8 gz8 = {0, 0, 0, 0, 0, 0, 0, 0};
      __builtin_amdgcn_tensor_load_to_lds(g0, g1, gz4, gz4, gz8, 0);
    }
#else
    for (int j = 0; j < 16; ++j) {                  // manual B staging fallback
      int e = j * 32 + lane;
      int kk = e >> 4, n = e & 15;
      int kb = k0 + kk, co = ntile * 16 + n;
      _Float16 w = (_Float16)0.0f;
      if (kb < K && co < COUT) w = wp[(long)kb * COUT + co];
      Bs[e] = w;
    }
#endif

    // ---- Stage A: gather 32 voxels for this (tap, ci)
    #pragma unroll 4
    for (int j = 0; j < 32; ++j) {
      int b = vb[j], oz = vz[j], oy = vy[j], ox = vx[j];
      int iz, iy, ix; bool ok = kvalid;
      if (TRANSPOSED) {                 // lhs_dilation=2 adjoint conv
        int tz = oz - PADLO + kz; ok = ok && !(tz & 1) && tz >= 0; iz = tz >> 1;
        int ty = oy - PADLO + ky; ok = ok && !(ty & 1) && ty >= 0; iy = ty >> 1;
        int tx = ox - PADLO + kx; ok = ok && !(tx & 1) && tx >= 0; ix = tx >> 1;
      } else {
        iz = oz * STRIDE - PADLO + kz;
        iy = oy * STRIDE - PADLO + ky;
        ix = ox * STRIDE - PADLO + kx;
      }
      ok = ok && (unsigned)iz < (unsigned)DI && (unsigned)iy < (unsigned)DI
              && (unsigned)ix < (unsigned)DI;
      float f = 0.0f;
      if (ok) {
        long vin = (((long)b * DI + iz) * DI + iy) * DI + ix;
        if (CIN2 == 0 || ci < CIN1) {
          f = (float)in1[vin * CIN1 + ci];
          if (IN_RELU) f = fmaxf(f, 0.0f);          // relu(skip) encoder path
        } else {
          f = (float)in2[vin * CIN2 + (ci - CIN1)]; // fused concat input
        }
      }
      As[j * 32 + lane] = (_Float16)f;
    }
    __syncthreads();
#if USE_TDM
    __builtin_amdgcn_s_wait_tensorcnt(0);           // B tile landed in LDS
#endif

    // ---- Fragment gather per CDNA5 16-bit A(16x32)/B(32x16) layouts
    v16h a0, a1, bfr;
    #pragma unroll
    for (int v = 0; v < 8; ++v) {
      const int ka = (v >> 2) * 16 + half * 8 + (v & 3) * 2;
      a0[2 * v]     = As[lan16 * 32 + ka];
      a0[2 * v + 1] = As[lan16 * 32 + ka + 1];
      a1[2 * v]     = As[(lan16 + 16) * 32 + ka];
      a1[2 * v + 1] = As[(lan16 + 16) * 32 + ka + 1];
      const int kb = (v >> 2) * 16 + half * 8 + (v & 3) * 2;
      bfr[2 * v]     = Bs[kb * 16 + lan16];
      bfr[2 * v + 1] = Bs[(kb + 1) * 16 + lan16];
    }
    acc0 = __builtin_amdgcn_wmma_f32_16x16x32_f16(false, a0, false, bfr,
                                                  (short)0, acc0, false, false);
    acc1 = __builtin_amdgcn_wmma_f32_16x16x32_f16(false, a1, false, bfr,
                                                  (short)0, acc1, false, false);
    __syncthreads();
  }

  // ---- Epilogue: folded BN, optional ReLU, store
  const int  co  = ntile * 16 + lan16;
  const bool cok = co < COUT;
  float scale = 1.0f, bias = 0.0f;
  if (!FINAL && cok) {
    float s = bn_g[co] / sqrtf(bn_v[co] + 1e-5f);
    scale = s;
    bias  = bn_b[co] - bn_m[co] * s;
  }
  #pragma unroll
  for (int r = 0; r < 8; ++r) {
    #pragma unroll
    for (int g = 0; g < 2; ++g) {
      float f = (g == 0 ? acc0[r] : acc1[r]) * scale + bias;
      if (OUT_RELU) f = fmaxf(f, 0.0f);
      int vox = mtile * 32 + g * 16 + half * 8 + r;   // C/D VGPR layout
      if (vox < M && cok) {
        if (FINAL) {                                   // fp32 NCDHW output
          int xq = vox % DO; int t = vox / DO;
          int yq = t % DO;   t /= DO;
          int zq = t % DO;   int bq = t / DO;
          long o = ((((long)bq * COUT + co) * DO + zq) * DO + yq) * DO + xq;
          out_f32[o] = f;
        } else {                                       // f16 NDHWC
          out[(long)vox * COUT + co] = (_Float16)f;
        }
      }
    }
  }
}

// Repack OIDHW fp32 weights -> [tap*Cin][Cout] f16 (GEMM B layout).
template<int COUT, int CIN, int TAPS>
__global__ void pack_w(const float* __restrict__ w, _Float16* __restrict__ wp)
{
  constexpr int TOTAL = COUT * CIN * TAPS;
  int idx = blockIdx.x * blockDim.x + threadIdx.x;
  if (idx >= TOTAL) return;
  int co  = idx / (CIN * TAPS);
  int rem = idx - co * (CIN * TAPS);
  int ci  = rem / TAPS;
  int tap = rem - ci * TAPS;
  wp[(tap * CIN + ci) * COUT + co] = (_Float16)w[idx];
}

// fp32 NCDHW (2,4,96,96,96) -> f16 NDHWC
__global__ void x_to_ndhwc(const float* __restrict__ x, _Float16* __restrict__ o)
{
  constexpr long TOTAL = 2L * 4 * 96 * 96 * 96;
  long idx = (long)blockIdx.x * blockDim.x + threadIdx.x;
  if (idx >= TOTAL) return;
  int xw = (int)(idx % 96); long t = idx / 96;
  int yh = (int)(t % 96);   t /= 96;
  int zd = (int)(t % 96);   t /= 96;
  int c  = (int)(t % 4);    int b = (int)(t / 4);
  o[((((long)b * 96 + zd) * 96 + yh) * 96 + xw) * 4 + c] = (_Float16)x[idx];
}

extern "C" void kernel_launch(void* const* d_in, const int* in_sizes, int n_in,
                              void* d_out, int out_size, void* d_ws, size_t ws_size,
                              hipStream_t stream)
{
  (void)in_sizes; (void)n_in; (void)out_size; (void)ws_size;

  char*  ws  = (char*)d_ws;
  size_t off = 0;
  auto alloc16 = [&](long elems) -> _Float16* {
    _Float16* p = (_Float16*)(ws + off);
    off += ((size_t)elems * 2 + 255) / 256 * 256;
    return p;
  };

#define PACK(IDX, CO, CI, TP, NAME)                                            \
  _Float16* NAME = alloc16((long)(CO) * (CI) * (TP));                          \
  pack_w<CO, CI, TP><<<dim3(((CO) * (CI) * (TP) + 255) / 256), dim3(256), 0,   \
                       stream>>>((const float*)d_in[IDX], NAME)

#define CONV(DI, DO, C1, C2, CO, KS, ST, PL, TR, IR, OR, FIN, IN1, IN2, WPP,   \
             BNI, OUTP, OUTF)                                                  \
  do {                                                                         \
    dim3 g((2 * (DO) * (DO) * (DO) + 31) / 32, ((CO) + 15) / 16);              \
    conv3d_wmma<DI, DO, C1, C2, CO, KS, ST, PL, TR, IR, OR, FIN>               \
        <<<g, dim3(32), 0, stream>>>(                                          \
            IN1, IN2, WPP,                                                     \
            (BNI) >= 0 ? (const float*)d_in[BNI] : nullptr,                    \
            (BNI) >= 0 ? (const float*)d_in[(BNI) + 1] : nullptr,              \
            (BNI) >= 0 ? (const float*)d_in[(BNI) + 2] : nullptr,              \
            (BNI) >= 0 ? (const float*)d_in[(BNI) + 3] : nullptr,              \
            OUTP, OUTF);                                                       \
  } while (0)

  const long V0 = 2L * 96 * 96 * 96, V1 = 2L * 48 * 48 * 48,
             V2 = 2L * 24 * 24 * 24, V3 = 2L * 12 * 12 * 12, V4 = 2L * 6 * 6 * 6;

  // Input convert: NCDHW fp32 -> NDHWC f16
  _Float16* ax = alloc16(V0 * 4);
  x_to_ndhwc<<<dim3((unsigned)((V0 * 4 + 255) / 256)), dim3(256), 0, stream>>>(
      (const float*)d_in[0], ax);

  // Encoder
  PACK(1,  8,   4,  27, wp_c0); _Float16* s0  = alloc16(V0 * 8);
  CONV(96, 96, 4,  0, 8,   3, 1, 1, 0, 0, 0, 0, ax,  nullptr, wp_c0, 2,  s0,  nullptr);
  PACK(6,  16,  8,  27, wp_1a); _Float16* a1a = alloc16(V1 * 16);
  CONV(96, 48, 8,  0, 16,  3, 2, 1, 0, 1, 1, 0, s0,  nullptr, wp_1a, 7,  a1a, nullptr);
  PACK(11, 16,  16, 27, wp_1b); _Float16* s1  = alloc16(V1 * 16);
  CONV(48, 48, 16, 0, 16,  3, 1, 1, 0, 0, 0, 0, a1a, nullptr, wp_1b, 12, s1,  nullptr);
  PACK(16, 32,  16, 27, wp_2a); _Float16* a2a = alloc16(V2 * 32);
  CONV(48, 24, 16, 0, 32,  3, 2, 1, 0, 1, 1, 0, s1,  nullptr, wp_2a, 17, a2a, nullptr);
  PACK(21, 32,  32, 27, wp_2b); _Float16* s2  = alloc16(V2 * 32);
  CONV(24, 24, 32, 0, 32,  3, 1, 1, 0, 0, 0, 0, a2a, nullptr, wp_2b, 22, s2,  nullptr);
  PACK(26, 64,  32, 27, wp_3a); _Float16* a3a = alloc16(V3 * 64);
  CONV(24, 12, 32, 0, 64,  3, 2, 1, 0, 1, 1, 0, s2,  nullptr, wp_3a, 27, a3a, nullptr);
  PACK(31, 64,  64, 27, wp_3b); _Float16* s3  = alloc16(V3 * 64);
  CONV(12, 12, 64, 0, 64,  3, 1, 1, 0, 0, 0, 0, a3a, nullptr, wp_3b, 32, s3,  nullptr);
  PACK(36, 128, 64, 27, wp_c4); _Float16* s4  = alloc16(V4 * 128);
  CONV(12, 6,  64, 0, 128, 3, 2, 1, 0, 1, 1, 0, s3,  nullptr, wp_c4, 37, s4,  nullptr);

  // Decoder (transpose convs; dual input = fused concat with raw skips)
  PACK(41, 64, 128, 27, wp_t4); _Float16* d4 = alloc16(V3 * 64);
  CONV(6,  12, 128, 0,  64, 3, 1, 1, 1, 0, 1, 0, s4, nullptr, wp_t4, 42, d4, nullptr);
  PACK(46, 64, 128, 27, wp_t3); _Float16* d3 = alloc16(V2 * 64);
  CONV(12, 24, 64,  64, 64, 3, 1, 1, 1, 0, 1, 0, d4, s3,      wp_t3, 47, d3, nullptr);
  PACK(51, 48, 96,  27, wp_t2); _Float16* d2 = alloc16(V1 * 48);
  CONV(24, 48, 64,  32, 48, 3, 1, 1, 1, 0, 1, 0, d3, s2,      wp_t2, 52, d2, nullptr);
  PACK(56, 24, 64,  27, wp_t1); _Float16* d1 = alloc16(V0 * 24);
  CONV(48, 96, 48,  16, 24, 3, 1, 1, 1, 0, 1, 0, d2, s1,      wp_t1, 57, d1, nullptr);

  // Final 1x1x1 conv -> fp32 NCDHW straight into d_out
  PACK(61, 2, 32, 1, wp_co);
  CONV(96, 96, 24, 8, 2, 1, 1, 0, 0, 0, 0, 1, d1, s0, wp_co, -1, nullptr,
       (float*)d_out);

#undef PACK
#undef CONV
}
